// TwoPathLowRankLinear_66700842107013
// MI455X (gfx1250) — matrix operand order repaired and checked by reading, not compiled, not run on hardware
//
#include <hip/hip_runtime.h>
#include <hip/hip_bf16.h>

// ---------------------------------------------------------------------------
// TwoPathLowRankLinear for MI455X (gfx1250, wave32, WMMA)
//   x [16384, 4096] fp32 ; ranks 64+64 concat -> R=128
//   out = (x Vc^T) Uc^T + bias   with Vc/Uc dequantized bf16 weights
// prep: int codes -> bf16 Vc[128,4096], Uc[4096,128] in ws
// down: Z = x * Vc^T   (double-buffered LDS, 4 WMMA / barrier)
// up:   out = Z * Uc^T + bias
// ---------------------------------------------------------------------------

#define DIMD 4096
#define RTOT 128
#define RHALF 64
#define TOKENS 16384
#define KCH 128            // K-chunk for the down kernel
#define NCH (DIMD / KCH)   // 32 chunks
#define LROW 40            // padded LDS row in halves (80B: bank-conflict-free)

typedef __bf16 v16bf __attribute__((ext_vector_type(16)));
typedef float  v8f   __attribute__((ext_vector_type(8)));
typedef unsigned int u32x4 __attribute__((ext_vector_type(4)));

__device__ __forceinline__ unsigned short f2bf(float f) {
  union { float f; unsigned u; } v; v.f = f;
  unsigned r = v.u + 0x7FFFu + ((v.u >> 16) & 1u);  // round-to-nearest-even
  return (unsigned short)(r >> 16);
}
__device__ __forceinline__ unsigned pack2bf(float a, float b) {
  return (unsigned)f2bf(a) | ((unsigned)f2bf(b) << 16);
}

// A-fragment K permutation for 16-bit 16x32 tiles (ISA 7.12.2):
// lane group g=(lane>>4) consumes K = {g*8..g*8+7, 16+g*8..16+g*8+7};
// permute K so each lane's 16 halves are 32B-contiguous. 8-aligned runs of K
// stay contiguous under this map.
__device__ __forceinline__ int permk(int k) {
  return (k & 7) | (((k >> 4) & 1) << 3) | (((k >> 3) & 1) << 4);
}

__device__ __forceinline__ v16bf load_frag(const unsigned short* p) {
  union { u32x4 q[2]; v16bf v; } u;
  const u32x4* s = reinterpret_cast<const u32x4*>(p);  // 16B-aligned by construction
  u.q[0] = s[0];
  u.q[1] = s[1];
  return u.v;
}

// ---------------------------------------------------------------------------
// Kernel 0: dequantize weights into bf16 Vc[128,4096] and Uc[4096,128]
// ---------------------------------------------------------------------------
__global__ __launch_bounds__(256) void tplr_prep(
    const int* __restrict__ uh_q, const float* __restrict__ uh_s,
    const int* __restrict__ vh_q, const float* __restrict__ vh_s,
    const int* __restrict__ ul_q, const float* __restrict__ ul_s,
    const int* __restrict__ vl_q, const float* __restrict__ vl_s,
    unsigned short* __restrict__ Vc, unsigned short* __restrict__ Uc) {
  const int NV = RTOT * DIMD;
  int i = blockIdx.x * 256 + threadIdx.x;
  if (i < NV) {
    int r = i >> 12, d = i & (DIMD - 1);
    float v = (r < RHALF)
                  ? (float)vh_q[(size_t)r * DIMD + d] * vh_s[r]
                  : (float)vl_q[(size_t)(r - RHALF) * DIMD + d] * vl_s[r - RHALF];
    Vc[i] = f2bf(v);
  } else {
    int j = i - NV;
    int o = j >> 7, r = j & (RTOT - 1);
    float v = (r < RHALF)
                  ? (float)uh_q[(size_t)o * RHALF + r] * uh_s[o]
                  : (float)ul_q[(size_t)o * RHALF + (r - RHALF)] * ul_s[o];
    Uc[j] = f2bf(v);
  }
}

// ---------------------------------------------------------------------------
// Kernel 1: Z[16384,128] = x * Vc^T
//   16 token rows / block; 8 waves = 8 rank tiles; K in chunks of 128 with
//   double-buffered LDS; one barrier and 4 WMMAs per chunk; next chunk's
//   global_load_b128 x2 issued before compute (latency hidden by WMMAs).
// ---------------------------------------------------------------------------
__global__ __launch_bounds__(256) void tplr_down(
    const float* __restrict__ x, const unsigned short* __restrict__ Vc,
    unsigned short* __restrict__ Zb) {
  __shared__ __align__(64) unsigned short At[2][4][16][LROW];

  const int tid   = threadIdx.x;
  const int lane  = tid & 31;
  const int wave  = tid >> 5;          // rank tile 0..7
  const int row0  = blockIdx.x * 16;
  const int n     = lane & 15;
  const int g     = lane >> 4;
  const int rbase = wave * 16;

  const unsigned short* Brow = Vc + (size_t)(rbase + n) * DIMD + g * 16;

  // loader: thread -> 8 consecutive k of one row (2 x float4 = 32B global)
  const int lm   = tid >> 4;           // row 0..15
  const int lk   = (tid & 15) * 8;     // k 0..120 step 8
  const int lks  = lk >> 5;            // ksub 0..3
  const int lpos = lk & 31;            // 8-aligned -> 8 contiguous LDS halves
  const float* xp = x + (size_t)(row0 + lm) * DIMD + lk;

  float4 r0, r1;
  // prologue: chunk 0
  r0 = *reinterpret_cast<const float4*>(xp);
  r1 = *reinterpret_cast<const float4*>(xp + 4);
  {
    u32x4 p;
    p.x = pack2bf(r0.x, r0.y); p.y = pack2bf(r0.z, r0.w);
    p.z = pack2bf(r1.x, r1.y); p.w = pack2bf(r1.z, r1.w);
    *reinterpret_cast<u32x4*>(&At[0][lks][lm][lpos]) = p;
  }
  __syncthreads();

  v8f c = {};
  for (int ch = 0; ch < NCH; ++ch) {
    if (ch + 1 < NCH) {                       // prefetch next chunk (32B/thread)
      r0 = *reinterpret_cast<const float4*>(xp + (ch + 1) * KCH);
      r1 = *reinterpret_cast<const float4*>(xp + (ch + 1) * KCH + 4);
    }
    const unsigned short(*buf)[16][LROW] = At[ch & 1];
#pragma unroll
    for (int s = 0; s < 4; ++s) {
      v16bf a = load_frag(&buf[s][n][g * 16]);          // 2x ds_load_b128
      v16bf b = load_frag(Brow + ch * KCH + s * 32);    // 2x global_load_b128 (L2)
      c = __builtin_amdgcn_wmma_f32_16x16x32_bf16(false, a, false, b,
                                                  (short)0, c, false, false);
    }
    if (ch + 1 < NCH) {                       // fill other buffer
      u32x4 p;
      p.x = pack2bf(r0.x, r0.y); p.y = pack2bf(r0.z, r0.w);
      p.z = pack2bf(r1.x, r1.y); p.w = pack2bf(r1.z, r1.w);
      *reinterpret_cast<u32x4*>(&At[(ch + 1) & 1][lks][lm][lpos]) = p;
    }
    __syncthreads();
  }

  // C/D layout: VGPR i -> row i + 8*g, col = n
#pragma unroll
  for (int i = 0; i < 8; ++i)
    Zb[(size_t)(row0 + 8 * g + i) * RTOT + rbase + n] = f2bf(c[i]);
}

// ---------------------------------------------------------------------------
// Kernel 2: out[16384,4096] = Z * Uc^T + bias  (16x128 out tile per block)
// ---------------------------------------------------------------------------
__global__ __launch_bounds__(256) void tplr_up(
    const unsigned short* __restrict__ Zb, const unsigned short* __restrict__ Uc,
    const float* __restrict__ bias, float* __restrict__ out) {
  __shared__ __align__(64) unsigned short At[4][16][LROW];

  const int tid  = threadIdx.x;
  const int lane = tid & 31;
  const int wave = tid >> 5;
  const int row0 = blockIdx.x * 16;
  const int n    = lane & 15;
  const int g    = lane >> 4;
  const int o0   = blockIdx.y * 128 + wave * 16;

  // stage Z[row0..+16, 0..128) in fragment order (2048 elems, 2/thread/iter)
#pragma unroll
  for (int it = 0; it < 4; ++it) {
    int e = (tid + it * 256) * 2;
    int m = e >> 7;
    int k = e & 127;  // even
    unsigned v = *reinterpret_cast<const unsigned*>(Zb + (size_t)(row0 + m) * RTOT + k);
    *reinterpret_cast<unsigned*>(&At[k >> 5][m][k & 31]) = v;
  }
  __syncthreads();

  const unsigned short* Brow = Uc + (size_t)(o0 + n) * RTOT + g * 16;

  v8f c = {};
#pragma unroll
  for (int kc = 0; kc < 4; ++kc) {
    v16bf a = load_frag(&At[kc][n][g * 16]);
    v16bf b = load_frag(Brow + kc * 32);
    c = __builtin_amdgcn_wmma_f32_16x16x32_bf16(false, a, false, b,
                                                (short)0, c, false, false);
  }

  float bv = bias[o0 + n];
#pragma unroll
  for (int i = 0; i < 8; ++i)
    out[(size_t)(row0 + 8 * g + i) * DIMD + o0 + n] = c[i] + bv;
}

// ---------------------------------------------------------------------------
extern "C" void kernel_launch(void* const* d_in, const int* in_sizes, int n_in,
                              void* d_out, int out_size, void* d_ws, size_t ws_size,
                              hipStream_t stream) {
  const float* x    = (const float*)d_in[0];
  const int*   uh_q = (const int*)d_in[1];
  const float* uh_s = (const float*)d_in[2];
  const int*   vh_q = (const int*)d_in[3];
  const float* vh_s = (const float*)d_in[4];
  const int*   ul_q = (const int*)d_in[5];
  const float* ul_s = (const float*)d_in[6];
  const int*   vl_q = (const int*)d_in[7];
  const float* vl_s = (const float*)d_in[8];
  const float* bias = (const float*)d_in[9];
  float* out = (float*)d_out;

  // workspace layout (bf16): Vc[128*4096] | Uc[4096*128] | Z[16384*128]  = 6 MB
  unsigned short* Vc = (unsigned short*)d_ws;
  unsigned short* Uc = Vc + (size_t)RTOT * DIMD;
  unsigned short* Zb = Uc + (size_t)DIMD * RTOT;

  tplr_prep<<<(2 * RTOT * DIMD) / 256, 256, 0, stream>>>(
      uh_q, uh_s, vh_q, vh_s, ul_q, ul_s, vl_q, vl_s, Vc, Uc);

  tplr_down<<<TOKENS / 16, 256, 0, stream>>>(x, Vc, Zb);

  dim3 g2(TOKENS / 16, DIMD / 128);
  tplr_up<<<g2, 256, 0, stream>>>(Zb, Uc, bias, out);
}
